// NetGCN3Layer_61263413510541
// MI455X (gfx1250) — matrix operand order repaired and checked by reading, not compiled, not run on hardware
//
#include <hip/hip_runtime.h>

// ---------------------------------------------------------------------------
// Types for CDNA5 WMMA (wave32): v_wmma_f32_16x16x32_bf16
// ---------------------------------------------------------------------------
typedef __bf16 bf16;
typedef __attribute__((ext_vector_type(16))) __bf16 v16bf;
typedef __attribute__((ext_vector_type(8)))  float  v8f;

__device__ __forceinline__ bf16 f2bf(float f) {
  union { float f; unsigned u; } c; c.f = f;
  unsigned r = c.u + 0x7fffu + ((c.u >> 16) & 1u);   // round-to-nearest-even
  union { unsigned short s; bf16 b; } o; o.s = (unsigned short)(r >> 16);
  return o.b;
}
__device__ __forceinline__ bf16 bfzero() {
  union { unsigned short s; bf16 b; } o; o.s = 0; return o.b;
}

// Async copy: cache -> LDS without VGPR round trip (CDNA5, tracked by ASYNCcnt)
__device__ __forceinline__ void async_g2l_b32(unsigned ldsAddr, const unsigned* gaddr) {
  asm volatile("global_load_async_to_lds_b32 %0, %1, off"
               :: "v"(ldsAddr), "v"(gaddr) : "memory");
}
__device__ __forceinline__ void wait_async() {
  asm volatile("s_wait_asynccnt 0x0" ::: "memory");
}

// Network constants
#define BATCH 128
#define NV1 4096
#define KCH 25     // Chebyshev order
#define DEG 16     // fixed sparse degree

// ---------------------------------------------------------------------------
// x (B, 4096, 1) -> x0 (V, B) fp32 ring slot + bf16 basis slice 0
// ---------------------------------------------------------------------------
__global__ void k_init_x0(const float* __restrict__ x,
                          float* __restrict__ r0, bf16* __restrict__ xb0) {
  int idx = blockIdx.x * blockDim.x + threadIdx.x;        // idx = v*128 + b
  if (idx >= NV1 * BATCH) return;
  int v = idx >> 7, b = idx & 127;
  float val = x[(size_t)b * NV1 + v];
  r0[idx]  = val;
  xb0[idx] = f2bf(val);
}

// ---------------------------------------------------------------------------
// SpMM step of Chebyshev recursion: out = scale*(L @ xin) + beta*xsub
//   L: fixed degree-16 rows; xin/xsub/out: (V, M) fp32.
//   Also mirrors the result into the bf16 basis slice (GEMM input).
// Grid: (V, M/128), block 128. Edge list staged through LDS.
// ---------------------------------------------------------------------------
__global__ void k_spmm(const int* __restrict__ cols, const float* __restrict__ vals,
                       const float* __restrict__ xin, const float* __restrict__ xsub,
                       float scale, float beta,
                       float* __restrict__ outf, bf16* __restrict__ outb, int M) {
  __shared__ int   sc[DEG];
  __shared__ float sv[DEG];
  const int v = blockIdx.x;
  const int t = threadIdx.x;
  if (t < DEG) { sc[t] = cols[v * DEG + t]; sv[t] = vals[v * DEG + t]; }
  __syncthreads();
  const int m = blockIdx.y * 128 + t;
  float acc = 0.f;
#pragma unroll
  for (int j = 0; j < DEG; ++j)
    acc = fmaf(sv[j], xin[(size_t)sc[j] * M + m], acc);
  size_t oi = (size_t)v * M + m;
  float r = fmaf(beta, xsub[oi], scale * acc);
  outf[oi] = r;
  outb[oi] = f2bf(r);
}

// ---------------------------------------------------------------------------
// Pre-fragment weights into the WMMA *A-operand* lane layout (transposed GEMM:
// A[o,k] = W[kb*KV + k, o]).  Record (kb, oTile): lane L = (hi,lo) holds row
// o = oTile*16+lo; element e maps to K = hi*16 + e
// (lanes<16 hold K{0-15}, lanes>=16 hold K{16-31}; matches B fragments).
// K-locals >= KV zero-padded (cheb: 25 -> 32).
// ---------------------------------------------------------------------------
__global__ void k_prep_afrag(const float* __restrict__ W, bf16* __restrict__ dst,
                             int KV, int nKB, int nCT, int Cout) {
  long total = (long)nKB * nCT * 512;                     // 32 lanes * 16 elems
  long idx = (long)blockIdx.x * blockDim.x + threadIdx.x;
  if (idx >= total) return;
  int  e    = (int)(idx & 15);
  int  lane = (int)((idx >> 4) & 31);
  long rec  = idx >> 9;
  int  ctG  = (int)(rec % nCT);
  long kb   = rec / nCT;
  int  lo = lane & 15, hi = lane >> 4;
  int  klocal = hi * 16 + e;
  int  col    = ctG * 16 + lo;                            // output feature o
  float val = 0.f;
  if (klocal < KV) val = W[((size_t)kb * KV + klocal) * Cout + col];
  dst[idx] = f2bf(val);
}

// ---------------------------------------------------------------------------
// Chebyshev projection GEMM (transposed):
//   out[v, o, b] = relu( sum_f sum_k W[f*25+k, o] * Xb[k][v*M + f*128 + b] + bias[o] )
// One block per graph node v (8 waves = 8 b-tiles of 16).  Per feature f the
// block async-copies the (25 x 128) basis panel into LDS (cache->LDS DMA,
// no VGPR round trip), builds B-fragments from LDS, A-fragments are
// contiguous 32B weight records.  Output layout (V, Fout, 128).
// ---------------------------------------------------------------------------
template <int NCT>   // Fout = NCT*16
__global__ void k_cheb_gemm(const bf16* __restrict__ Xb, const bf16* __restrict__ WfragA,
                            const float* __restrict__ bias,
                            float* __restrict__ outf, bf16* __restrict__ outb,
                            int V, int M, int Fin, int relu) {
  const int Fout = NCT * 16;
  __shared__ unsigned shU[KCH * 64];                      // [k][128 bf16]
  const bf16* sh = (const bf16*)shU;
  const int t = threadIdx.x;
  const int lane = t & 31;
  const int lo = lane & 15, hi = lane >> 4;
  const int bBase = (t >> 5) * 16;                        // wave -> b-tile
  const int v = blockIdx.x;
  const size_t SSu = ((size_t)V * M) >> 1;                // slice stride in u32

  v8f acc[NCT];
#pragma unroll
  for (int ct = 0; ct < NCT; ++ct) acc[ct] = v8f{0.f,0.f,0.f,0.f,0.f,0.f,0.f,0.f};
  const bf16 z = bfzero();

  for (int f = 0; f < Fin; ++f) {
    const unsigned* gp = (const unsigned*)(Xb + (size_t)v * M + (size_t)f * BATCH);
    __syncthreads();                                      // LDS reuse from prev iter
    for (int i = t; i < KCH * 64; i += 256) {
      int k = i >> 6, w = i & 63;
      const unsigned* ga = gp + (size_t)k * SSu + w;
      async_g2l_b32((unsigned)(size_t)&shU[i], ga);
      // prefetch same element of the next feature panel (256B ahead)
      __builtin_prefetch((const char*)ga + 256, 0, 1);
    }
    wait_async();
    __syncthreads();

    v16bf bfr;                                            // B: K = hi*16 + e
#pragma unroll
    for (int e = 0; e < 16; ++e) {
      int k = hi * 16 + e;
      bfr[e] = (k < KCH) ? sh[k * BATCH + bBase + lo] : z;
    }
#pragma unroll
    for (int ct = 0; ct < NCT; ++ct) {
      v16bf aw = *(const v16bf*)(WfragA + (((size_t)f * NCT + ct) * 32 + lane) * 16);
      acc[ct] = __builtin_amdgcn_wmma_f32_16x16x32_bf16(
          false, aw, false, bfr, (short)0, acc[ct], false, false);
    }
  }
#pragma unroll
  for (int ct = 0; ct < NCT; ++ct) {
#pragma unroll
    for (int i = 0; i < 8; ++i) {
      int o = ct * 16 + i + hi * 8;                       // D row
      float val = acc[ct][i] + bias[o];
      if (relu) val = fmaxf(val, 0.f);
      size_t oi = ((size_t)v * Fout + o) * BATCH + bBase + lo;
      if (outf) outf[oi] = val;
      if (outb) outb[oi] = f2bf(val);
    }
  }
}

// ---------------------------------------------------------------------------
// pool4 + transpose: h (Vin, F, 128) -> x0_next (Vin/4, F*128) fp32 + bf16 slice0
// Fully coalesced on both sides.
// ---------------------------------------------------------------------------
__global__ void k_pool4t(const float* __restrict__ h, float* __restrict__ r0,
                         bf16* __restrict__ xb0, int Vin, int F) {
  int FB = F * BATCH;
  long total = (long)(Vin / 4) * FB;
  long idx = (long)blockIdx.x * blockDim.x + threadIdx.x;
  if (idx >= total) return;
  int v2 = (int)(idx / FB);
  int rem = (int)(idx - (long)v2 * FB);                   // rem = f*128 + b
  const float* hp = h + (size_t)(4 * v2) * FB + rem;
  float m = hp[0];
  m = fmaxf(m, hp[(size_t)FB]);
  m = fmaxf(m, hp[(size_t)2 * FB]);
  m = fmaxf(m, hp[(size_t)3 * FB]);
  r0[idx]  = m;
  xb0[idx] = f2bf(m);
}

// ---------------------------------------------------------------------------
// FC1 (transposed): out[o, b] = sum_k W[k, o] * h3[k, b] + bias[o]
//   h3 bf16 laid out (16384, 128) by layer-3 GEMM.  One block per 64-wide
//   output chunk; per 32-k block the activation panel (32x128) is async-
//   copied into LDS.  Output (512, 128) fp32.
// ---------------------------------------------------------------------------
__global__ void k_fc1_gemm(const bf16* __restrict__ Ab, const bf16* __restrict__ WfragA,
                           const float* __restrict__ bias, float* __restrict__ out) {
  __shared__ unsigned shU[32 * 64];                       // [k][128 bf16]
  const bf16* sh = (const bf16*)shU;
  const int t = threadIdx.x;
  const int lane = t & 31;
  const int lo = lane & 15, hi = lane >> 4;
  const int bBase = (t >> 5) * 16;
  const int oChunk = blockIdx.x;                          // 8 blocks

  v8f acc[4];
#pragma unroll
  for (int ct = 0; ct < 4; ++ct) acc[ct] = v8f{0.f,0.f,0.f,0.f,0.f,0.f,0.f,0.f};

  for (int kb = 0; kb < 512; ++kb) {
    const unsigned* gp = (const unsigned*)(Ab + (size_t)kb * 32 * BATCH);
    __syncthreads();
    for (int i = t; i < 32 * 64; i += 256) {
      async_g2l_b32((unsigned)(size_t)&shU[i], gp + i);
      __builtin_prefetch((const char*)(gp + i) + 8192, 0, 1);  // next k-panel
    }
    wait_async();
    __syncthreads();

    v16bf bfr;
#pragma unroll
    for (int e = 0; e < 16; ++e) {
      int k = hi * 16 + e;
      bfr[e] = sh[k * BATCH + bBase + lo];
    }
#pragma unroll
    for (int ct = 0; ct < 4; ++ct) {
      int ctG = oChunk * 4 + ct;
      v16bf aw = *(const v16bf*)(WfragA + (((size_t)kb * 32 + ctG) * 32 + lane) * 16);
      acc[ct] = __builtin_amdgcn_wmma_f32_16x16x32_bf16(
          false, aw, false, bfr, (short)0, acc[ct], false, false);
    }
  }
#pragma unroll
  for (int ct = 0; ct < 4; ++ct) {
#pragma unroll
    for (int i = 0; i < 8; ++i) {
      int o = oChunk * 64 + ct * 16 + i + hi * 8;
      out[(size_t)o * BATCH + bBase + lo] = acc[ct][i] + bias[o];
    }
  }
}

// ---------------------------------------------------------------------------
// BatchNorm (training stats over B=128, biased var) + ReLU.
// h laid out (512, 128): each feature's batch values are contiguous.
// ---------------------------------------------------------------------------
__global__ void k_bn(const float* __restrict__ h, const float* __restrict__ gamma,
                     const float* __restrict__ beta, float* __restrict__ out) {
  int col = blockIdx.x * blockDim.x + threadIdx.x;
  if (col >= 512) return;
  const float* hp = h + (size_t)col * BATCH;
  float s = 0.f, ss = 0.f;
  for (int b = 0; b < BATCH; ++b) { float t = hp[b]; s += t; ss += t * t; }
  float mean = s * (1.f / BATCH);
  float var  = ss * (1.f / BATCH) - mean * mean;
  float inv  = rsqrtf(var + 1e-5f);
  float g  = gamma[col] * inv;
  float bb = beta[col] - mean * g;
  float* op = out + (size_t)col * BATCH;
  for (int b = 0; b < BATCH; ++b)
    op[b] = fmaxf(fmaf(hp[b], g, bb), 0.f);
}

// ---------------------------------------------------------------------------
// FC2: logits[b,o] = sum_k hbn[k,b] * w[k,o] + bias[o]   (hbn is (512,128))
// ---------------------------------------------------------------------------
__global__ void k_fc2(const float* __restrict__ h, const float* __restrict__ w,
                      const float* __restrict__ bias, float* __restrict__ out) {
  int idx = blockIdx.x * blockDim.x + threadIdx.x;
  if (idx >= BATCH * 10) return;
  int b = idx / 10, o = idx % 10;
  float acc = bias[o];
  for (int k = 0; k < 512; ++k)
    acc = fmaf(h[(size_t)k * BATCH + b], w[k * 10 + o], acc);
  out[idx] = acc;
}

// ---------------------------------------------------------------------------
// In-place log_softmax over rows of (128, 10)
// ---------------------------------------------------------------------------
__global__ void k_lsm(float* __restrict__ out) {
  int b = threadIdx.x;   // 128 threads
  float x[10];
  float m = -1e30f;
#pragma unroll
  for (int o = 0; o < 10; ++o) { x[o] = out[b * 10 + o]; m = fmaxf(m, x[o]); }
  float s = 0.f;
#pragma unroll
  for (int o = 0; o < 10; ++o) s += expf(x[o] - m);
  float l = logf(s) + m;
#pragma unroll
  for (int o = 0; o < 10; ++o) out[b * 10 + o] = x[o] - l;
}

// ---------------------------------------------------------------------------
// Host launcher
// ---------------------------------------------------------------------------
extern "C" void kernel_launch(void* const* d_in, const int* in_sizes, int n_in,
                              void* d_out, int out_size, void* d_ws, size_t ws_size,
                              hipStream_t stream) {
  (void)in_sizes; (void)n_in; (void)out_size; (void)ws_size;
  const float* x    = (const float*)d_in[0];
  const int*   l0c  = (const int*)  d_in[2];
  const float* l0v  = (const float*)d_in[3];
  const int*   l2c  = (const int*)  d_in[5];
  const float* l2v  = (const float*)d_in[6];
  const int*   l4c  = (const int*)  d_in[8];
  const float* l4v  = (const float*)d_in[9];
  const float* w1   = (const float*)d_in[10];
  const float* b1   = (const float*)d_in[11];
  const float* w2   = (const float*)d_in[12];
  const float* b2   = (const float*)d_in[13];
  const float* w3   = (const float*)d_in[14];
  const float* b3   = (const float*)d_in[15];
  const float* fc1w = (const float*)d_in[16];
  const float* fc1b = (const float*)d_in[17];
  const float* bng  = (const float*)d_in[18];
  const float* bnb  = (const float*)d_in[19];
  const float* fc2w = (const float*)d_in[20];
  const float* fc2b = (const float*)d_in[21];
  float* out = (float*)d_out;

  // Workspace carving (every buffer fully written before read)
  char* ws = (char*)d_ws;
  size_t off = 0;
  auto alloc = [&](size_t bytes) -> char* {
    char* p = ws + off;
    off = (off + bytes + 255) & ~(size_t)255;
    return p;
  };
  const size_t VMMAX = 4194304;                          // max V*M (layer 2)
  float* ring0 = (float*)alloc(VMMAX * 4);
  float* ring1 = (float*)alloc(VMMAX * 4);
  float* ring2 = (float*)alloc(VMMAX * 4);
  float* ring[3] = { ring0, ring1, ring2 };
  bf16*  Xb    = (bf16*) alloc((size_t)KCH * VMMAX * 2); // bf16 Chebyshev basis
  float* hbuf  = (float*)alloc(16777216ull * 4);         // pre-pool activations (V,F,B)
  bf16*  h3bf  = (bf16*) alloc(2097152ull * 2);          // (16384, 128) bf16
  bf16*  wf1   = (bf16*) alloc(1024ull * 2);
  bf16*  wf2   = (bf16*) alloc(65536ull * 2);
  bf16*  wf3   = (bf16*) alloc(131072ull * 2);
  bf16*  wfc1  = (bf16*) alloc(8388608ull * 2);
  float* fc1h  = (float*)alloc(65536ull * 4);            // (512, 128)
  float* hbn   = (float*)alloc(65536ull * 4);            // (512, 128)

  // --- weight pre-fragmentation (bf16, WMMA A-operand layout) ---
  k_prep_afrag<<<4,     256, 0, stream>>>(w1,   wf1,  25, 1,   2,  32);
  k_prep_afrag<<<256,   256, 0, stream>>>(w2,   wf2,  25, 32,  4,  64);
  k_prep_afrag<<<512,   256, 0, stream>>>(w3,   wf3,  25, 64,  4,  64);
  k_prep_afrag<<<32768, 256, 0, stream>>>(fc1w, wfc1, 32, 512, 32, 512);

  // ---------------- Layer 1: V=4096, M=128, Fin=1, Fout=32 ----------------
  const size_t VM1 = (size_t)4096 * 128;
  k_init_x0<<<2048, 256, 0, stream>>>(x, ring[0], Xb);
  for (int k = 1; k < KCH; ++k) {
    k_spmm<<<dim3(4096, 1), 128, 0, stream>>>(
        l0c, l0v, ring[(k - 1) % 3], (k >= 2) ? ring[(k - 2) % 3] : ring[0],
        (k == 1) ? 1.f : 2.f, (k == 1) ? 0.f : -1.f,
        ring[k % 3], Xb + (size_t)k * VM1, 128);
  }
  k_cheb_gemm<2><<<4096, 256, 0, stream>>>(Xb, wf1, b1, hbuf, nullptr,
                                           4096, 128, 1, 1);
  k_pool4t<<<16384, 256, 0, stream>>>(hbuf, ring[0], Xb, 4096, 32);

  // ---------------- Layer 2: V=1024, M=4096, Fin=32, Fout=64 --------------
  const size_t VM2 = (size_t)1024 * 4096;
  for (int k = 1; k < KCH; ++k) {
    k_spmm<<<dim3(1024, 32), 128, 0, stream>>>(
        l2c, l2v, ring[(k - 1) % 3], (k >= 2) ? ring[(k - 2) % 3] : ring[0],
        (k == 1) ? 1.f : 2.f, (k == 1) ? 0.f : -1.f,
        ring[k % 3], Xb + (size_t)k * VM2, 4096);
  }
  k_cheb_gemm<4><<<1024, 256, 0, stream>>>(Xb, wf2, b2, hbuf, nullptr,
                                           1024, 4096, 32, 1);
  k_pool4t<<<8192, 256, 0, stream>>>(hbuf, ring[0], Xb, 1024, 64);

  // ---------------- Layer 3: V=256, M=8192, Fin=64, Fout=64 ---------------
  const size_t VM3 = (size_t)256 * 8192;
  for (int k = 1; k < KCH; ++k) {
    k_spmm<<<dim3(256, 64), 128, 0, stream>>>(
        l4c, l4v, ring[(k - 1) % 3], (k >= 2) ? ring[(k - 2) % 3] : ring[0],
        (k == 1) ? 1.f : 2.f, (k == 1) ? 0.f : -1.f,
        ring[k % 3], Xb + (size_t)k * VM3, 8192);
  }
  k_cheb_gemm<4><<<256, 256, 0, stream>>>(Xb, wf3, b3, nullptr, h3bf,
                                          256, 8192, 64, 1);

  // ---------------- Head: FC1 -> BN -> ReLU -> FC2 -> log_softmax ---------
  k_fc1_gemm<<<8, 256, 0, stream>>>(h3bf, wfc1, fc1b, fc1h);
  k_bn<<<2, 256, 0, stream>>>(fc1h, bng, bnb, hbn);
  k_fc2<<<5, 256, 0, stream>>>(hbn, fc2w, fc2b, out);
  k_lsm<<<1, 128, 0, stream>>>(out);
}
